// T5EncoderSelfAttentionFlex_39762807226673
// MI455X (gfx1250) — matrix operand order repaired and checked by reading, not compiled, not run on hardware
//
#include <hip/hip_runtime.h>
#include <hip/hip_bf16.h>

// ---------------------------------------------------------------------------
// T5 encoder self-attention (B=2,S=2048,D=1024,H=16,HD=64) for gfx1250.
// All GEMM work goes through v_wmma_f32_16x16x32_bf16 (wave32 WMMA).
// GEMMs: 32x64 tile per wave, double-buffered K loop.
// Attention: flash-style, fp32 bias/softmax, bias prefetch (global_prefetch).
// ---------------------------------------------------------------------------

#define B_ 2
#define S_ 2048
#define D_ 1024
#define H_ 16
#define HD_ 64
#define O3_ 3072           // 3*H*HD
#define NEGV (-1.0e9f)

typedef __attribute__((ext_vector_type(16))) __bf16 v16bf;
typedef __attribute__((ext_vector_type(8)))  float  v8f;

static __device__ __forceinline__ __bf16 f2bf(float f) {
    union { float f; unsigned int u; } v; v.f = f;
    unsigned int r = v.u + 0x7FFFu + ((v.u >> 16) & 1u);   // RNE
    unsigned short h = (unsigned short)(r >> 16);
    return __builtin_bit_cast(__bf16, h);
}

// A-fragment (16x32 bf16): lane holds row, two 8-elem contiguous K runs.
static __device__ __forceinline__ void loadA(v16bf& a, const __bf16* __restrict__ row,
                                             int kk, int half) {
    #pragma unroll
    for (int e = 0; e < 8; e++) a[e]     = row[kk + half * 8 + e];
    #pragma unroll
    for (int e = 0; e < 8; e++) a[8 + e] = row[kk + 16 + half * 8 + e];
}
// B-fragment (32x16 bf16): lane holds column, two 8-elem contiguous K runs.
static __device__ __forceinline__ void loadB(v16bf& b, const __bf16* __restrict__ col,
                                             int kk, int half) {
    #pragma unroll
    for (int e = 0; e < 8; e++) b[e]     = col[kk + half * 8 + e];
    #pragma unroll
    for (int e = 0; e < 8; e++) b[8 + e] = col[kk + 16 + half * 8 + e];
}

// fp32 -> bf16 elementwise conversion
__global__ __launch_bounds__(256) void cvt_bf16(const float* __restrict__ in,
                                                __bf16* __restrict__ out, int n) {
    int i = blockIdx.x * 256 + threadIdx.x;
    if (i < n) out[i] = f2bf(in[i]);
}

// ---------------------------------------------------------------------------
// Kernel 1: QKV projection.  out[s,o] = sum_d x[s,d] * Wqkv[o,d]
// One wave computes a 32(M) x 64(N) tile, K=1024, double-buffered by 32.
// q scaled by HD^-0.5 at write.  q,k: [B,H,S,HD] bf16; vT: [B,H,HD,S] bf16.
// ---------------------------------------------------------------------------
__global__ __launch_bounds__(128) void qkv_gemm(const __bf16* __restrict__ xb,
                                                const __bf16* __restrict__ wb,
                                                __bf16* __restrict__ qo,
                                                __bf16* __restrict__ ko,
                                                __bf16* __restrict__ vto) {
    const int lane = threadIdx.x & 31, wid = threadIdx.x >> 5;
    const int task = blockIdx.x * 4 + wid;           // 128 * 48 = 6144 tasks
    const int NT4  = O3_ / 64;                       // 48
    const int mt   = task / NT4;
    const int nt4  = task % NT4;
    const int half = lane >> 4, l16 = lane & 15;

    const __bf16* arow0 = xb + (size_t)(mt * 32 + l16) * D_;
    const __bf16* arow1 = xb + (size_t)(mt * 32 + 16 + l16) * D_;
    const __bf16* brow[4];
    #pragma unroll
    for (int t = 0; t < 4; t++)
        brow[t] = wb + (size_t)(nt4 * 64 + t * 16 + l16) * D_;

    v8f acc[2][4];
    #pragma unroll
    for (int m = 0; m < 2; m++)
        #pragma unroll
        for (int t = 0; t < 4; t++)
            #pragma unroll
            for (int i = 0; i < 8; i++) acc[m][t][i] = 0.0f;

    v16bf A0[2], A1[2], Bf0[4], Bf1[4];
    loadA(A0[0], arow0, 0, half); loadA(A0[1], arow1, 0, half);
    #pragma unroll
    for (int t = 0; t < 4; t++) loadB(Bf0[t], brow[t], 0, half);

    for (int kk = 0; kk < D_; kk += 64) {
        // stage next chunk (kk+32) while multiplying kk
        loadA(A1[0], arow0, kk + 32, half); loadA(A1[1], arow1, kk + 32, half);
        #pragma unroll
        for (int t = 0; t < 4; t++) loadB(Bf1[t], brow[t], kk + 32, half);
        #pragma unroll
        for (int t = 0; t < 4; t++) {
            acc[0][t] = __builtin_amdgcn_wmma_f32_16x16x32_bf16(
                false, A0[0], false, Bf0[t], (short)0, acc[0][t], false, false);
            acc[1][t] = __builtin_amdgcn_wmma_f32_16x16x32_bf16(
                false, A0[1], false, Bf0[t], (short)0, acc[1][t], false, false);
        }
        // stage chunk kk+64 (wraps harmlessly to 0 on last iter) while multiplying kk+32
        const int nk = (kk + 64) & (D_ - 1);
        loadA(A0[0], arow0, nk, half); loadA(A0[1], arow1, nk, half);
        #pragma unroll
        for (int t = 0; t < 4; t++) loadB(Bf0[t], brow[t], nk, half);
        #pragma unroll
        for (int t = 0; t < 4; t++) {
            acc[0][t] = __builtin_amdgcn_wmma_f32_16x16x32_bf16(
                false, A1[0], false, Bf1[t], (short)0, acc[0][t], false, false);
            acc[1][t] = __builtin_amdgcn_wmma_f32_16x16x32_bf16(
                false, A1[1], false, Bf1[t], (short)0, acc[1][t], false, false);
        }
    }

    const float scale = 0.125f;  // 64^-0.5
    #pragma unroll
    for (int t = 0; t < 4; t++) {
        int o    = nt4 * 64 + t * 16 + l16;
        int proj = o >> 10, rem = o & 1023;
        int h    = rem >> 6, d = rem & 63;
        #pragma unroll
        for (int m = 0; m < 2; m++) {
            #pragma unroll
            for (int r = 0; r < 8; r++) {
                int srow = mt * 32 + m * 16 + r + half * 8;
                int b    = srow >> 11, s = srow & (S_ - 1);
                float v  = acc[m][t][r];
                size_t bh = (size_t)(b * H_ + h);
                if (proj == 0)      qo [(bh * S_ + s) * HD_ + d] = f2bf(v * scale);
                else if (proj == 1) ko [(bh * S_ + s) * HD_ + d] = f2bf(v);
                else                vto[(bh * HD_ + d) * S_ + s] = f2bf(v);
            }
        }
    }
}

// ---------------------------------------------------------------------------
// Kernel 2: flash attention with T5 relative-position bias + key/query mask.
// One wave per (b,h,16-query tile); keys streamed in chunks of 32.
// Scores: 4 WMMAs; online softmax in fp32; V fragments hoisted above softmax;
// bias prefetched 2 chunks ahead; P re-laid-out via per-wave LDS tile
// (C-layout -> A-layout) as bf16; P*V: 4 WMMAs into rescaled acc.
// Writes a[B,S,H*HD] bf16.
// ---------------------------------------------------------------------------
__global__ __launch_bounds__(128) void attn(const __bf16* __restrict__ q,
                                            const __bf16* __restrict__ k,
                                            const __bf16* __restrict__ vt,
                                            const int* __restrict__ mask,
                                            const float* __restrict__ bias,
                                            __bf16* __restrict__ aout) {
    __shared__ float pbuf[4][16][36];
    const int lane = threadIdx.x & 31, wid = threadIdx.x >> 5;
    const int task = blockIdx.x * 4 + wid;           // 2*16*128 = 4096
    const int QT   = S_ / 16;                        // 128
    const int qt   = task % QT;
    const int h    = (task / QT) & (H_ - 1);
    const int b    = task / (QT * H_);
    const int half = lane >> 4, l16 = lane & 15;

    const __bf16* qb = q  + ((size_t)(b * H_ + h) * S_) * HD_;
    const __bf16* kb = k  + ((size_t)(b * H_ + h) * S_) * HD_;
    const __bf16* vb = vt + ((size_t)(b * H_ + h) * HD_) * S_;
    const float*  bb = bias + (size_t)h * S_ * S_;
    const int*    mr = mask + b * S_;

    // Q A-fragments (two 32-wide d chunks)
    v16bf aq[2];
    {
        const __bf16* qp = qb + (size_t)(qt * 16 + l16) * HD_;
        loadA(aq[0], qp, 0,  half);
        loadA(aq[1], qp, 32, half);
    }

    float m_i[8], l_i[8];
    v8f   oacc[4];
    int   mq[8];
    #pragma unroll
    for (int r = 0; r < 8; r++) {
        m_i[r] = -3.0e38f; l_i[r] = 0.0f;
        mq[r]  = mr[qt * 16 + r + half * 8];
    }
    #pragma unroll
    for (int t = 0; t < 4; t++)
        #pragma unroll
        for (int i = 0; i < 8; i++) oacc[t][i] = 0.0f;

    for (int kc = 0; kc < S_; kc += 32) {
        // --- scores: two 16x16 tiles, K=64 over two 32-wide chunks ---
        v8f s[2];
        #pragma unroll
        for (int t = 0; t < 2; t++) {
            const __bf16* kp = kb + (size_t)(kc + t * 16 + l16) * HD_;
            v8f st;
            #pragma unroll
            for (int i = 0; i < 8; i++) st[i] = 0.0f;
            #pragma unroll
            for (int c = 0; c < 2; c++) {
                v16bf bk;
                loadB(bk, kp, c * 32, half);
                st = __builtin_amdgcn_wmma_f32_16x16x32_bf16(
                    false, aq[c], false, bk, (short)0, st, false, false);
            }
            s[t] = st;
        }
        // --- hoist V B-fragments; latency hidden by softmax below ---
        v16bf bv[4];
        #pragma unroll
        for (int t = 0; t < 4; t++)
            loadB(bv[t], vb + (size_t)(t * 16 + l16) * S_, kc, half);
        // --- bias + mask (fp32); prefetch bias 2 chunks ahead ---
        #pragma unroll
        for (int t = 0; t < 2; t++) {
            int key = kc + t * 16 + l16;
            int mk  = mr[key];
            #pragma unroll
            for (int r = 0; r < 8; r++) {
                int qrow = qt * 16 + r + half * 8;
                float sv = s[t][r] + bb[(size_t)qrow * S_ + key];
                s[t][r]  = (mk && mq[r]) ? sv : NEGV;
                if (t == 0 && kc + 64 < S_)
                    __builtin_prefetch(&bb[(size_t)qrow * S_ + kc + 64 + l16], 0, 0);
            }
        }
        // --- online softmax: rows live in one 16-lane half ---
        #pragma unroll
        for (int r = 0; r < 8; r++) {
            float vmax = fmaxf(s[0][r], s[1][r]);
            #pragma unroll
            for (int off = 8; off >= 1; off >>= 1)
                vmax = fmaxf(vmax, __shfl_xor(vmax, off, 32));
            float mnew  = fmaxf(m_i[r], vmax);
            float alpha = __expf(m_i[r] - mnew);
            float p0 = __expf(s[0][r] - mnew);
            float p1 = __expf(s[1][r] - mnew);
            s[0][r] = p0; s[1][r] = p1;
            float rs = p0 + p1;
            #pragma unroll
            for (int off = 8; off >= 1; off >>= 1)
                rs += __shfl_xor(rs, off, 32);
            l_i[r] = l_i[r] * alpha + rs;
            m_i[r] = mnew;
            #pragma unroll
            for (int t = 0; t < 4; t++) oacc[t][r] *= alpha;
        }
        // --- C-layout P -> LDS -> A-layout bf16 ---
        #pragma unroll
        for (int t = 0; t < 2; t++)
            #pragma unroll
            for (int r = 0; r < 8; r++)
                pbuf[wid][r + half * 8][t * 16 + l16] = s[t][r];
        __syncthreads();
        v16bf pa;
        #pragma unroll
        for (int e = 0; e < 8; e++) pa[e]     = f2bf(pbuf[wid][l16][half * 8 + e]);
        #pragma unroll
        for (int e = 0; e < 8; e++) pa[8 + e] = f2bf(pbuf[wid][l16][16 + half * 8 + e]);
        // --- P*V: 4 WMMAs ---
        #pragma unroll
        for (int t = 0; t < 4; t++)
            oacc[t] = __builtin_amdgcn_wmma_f32_16x16x32_bf16(
                false, pa, false, bv[t], (short)0, oacc[t], false, false);
        __syncthreads();
    }

    // epilogue: normalize, zero fully-masked query rows, write [B,S,H*HD] bf16
    #pragma unroll
    for (int r = 0; r < 8; r++) {
        int qrow  = qt * 16 + r + half * 8;
        float inv = (mq[r] && l_i[r] > 0.0f) ? (1.0f / l_i[r]) : 0.0f;
        #pragma unroll
        for (int t = 0; t < 4; t++) {
            int f = h * HD_ + t * 16 + l16;
            aout[((size_t)(b * S_ + qrow)) * (H_ * HD_) + f] = f2bf(oacc[t][r] * inv);
        }
    }
}

// ---------------------------------------------------------------------------
// Kernel 3: output projection.  out[s,o] = sum_f a[s,f] * Wo[o,f], fp32 out.
// One wave computes a 32(M) x 64(N) tile, K=1024, double-buffered by 32.
// ---------------------------------------------------------------------------
__global__ __launch_bounds__(128) void oproj_gemm(const __bf16* __restrict__ ab,
                                                  const __bf16* __restrict__ wb,
                                                  float* __restrict__ out) {
    const int lane = threadIdx.x & 31, wid = threadIdx.x >> 5;
    const int task = blockIdx.x * 4 + wid;           // 128 * 16 = 2048 tasks
    const int NT4  = D_ / 64;                        // 16
    const int mt   = task / NT4;
    const int nt4  = task % NT4;
    const int half = lane >> 4, l16 = lane & 15;
    const int KD   = H_ * HD_;                       // 1024

    const __bf16* arow0 = ab + (size_t)(mt * 32 + l16) * KD;
    const __bf16* arow1 = ab + (size_t)(mt * 32 + 16 + l16) * KD;
    const __bf16* brow[4];
    #pragma unroll
    for (int t = 0; t < 4; t++)
        brow[t] = wb + (size_t)(nt4 * 64 + t * 16 + l16) * KD;

    v8f acc[2][4];
    #pragma unroll
    for (int m = 0; m < 2; m++)
        #pragma unroll
        for (int t = 0; t < 4; t++)
            #pragma unroll
            for (int i = 0; i < 8; i++) acc[m][t][i] = 0.0f;

    v16bf A0[2], A1[2], Bf0[4], Bf1[4];
    loadA(A0[0], arow0, 0, half); loadA(A0[1], arow1, 0, half);
    #pragma unroll
    for (int t = 0; t < 4; t++) loadB(Bf0[t], brow[t], 0, half);

    for (int kk = 0; kk < KD; kk += 64) {
        loadA(A1[0], arow0, kk + 32, half); loadA(A1[1], arow1, kk + 32, half);
        #pragma unroll
        for (int t = 0; t < 4; t++) loadB(Bf1[t], brow[t], kk + 32, half);
        #pragma unroll
        for (int t = 0; t < 4; t++) {
            acc[0][t] = __builtin_amdgcn_wmma_f32_16x16x32_bf16(
                false, A0[0], false, Bf0[t], (short)0, acc[0][t], false, false);
            acc[1][t] = __builtin_amdgcn_wmma_f32_16x16x32_bf16(
                false, A0[1], false, Bf0[t], (short)0, acc[1][t], false, false);
        }
        const int nk = (kk + 64) & (KD - 1);
        loadA(A0[0], arow0, nk, half); loadA(A0[1], arow1, nk, half);
        #pragma unroll
        for (int t = 0; t < 4; t++) loadB(Bf0[t], brow[t], nk, half);
        #pragma unroll
        for (int t = 0; t < 4; t++) {
            acc[0][t] = __builtin_amdgcn_wmma_f32_16x16x32_bf16(
                false, A1[0], false, Bf1[t], (short)0, acc[0][t], false, false);
            acc[1][t] = __builtin_amdgcn_wmma_f32_16x16x32_bf16(
                false, A1[1], false, Bf1[t], (short)0, acc[1][t], false, false);
        }
    }

    #pragma unroll
    for (int t = 0; t < 4; t++) {
        int o = nt4 * 64 + t * 16 + l16;
        #pragma unroll
        for (int m = 0; m < 2; m++) {
            #pragma unroll
            for (int r = 0; r < 8; r++) {
                int srow = mt * 32 + m * 16 + r + half * 8;
                out[(size_t)srow * D_ + o] = acc[m][t][r];
            }
        }
    }
}

// ---------------------------------------------------------------------------
extern "C" void kernel_launch(void* const* d_in, const int* in_sizes, int n_in,
                              void* d_out, int out_size, void* d_ws, size_t ws_size,
                              hipStream_t stream) {
    const float* x    = (const float*)d_in[0];   // [B,S,D]
    const int*   mask = (const int*)  d_in[1];   // [B,S]
    const float* wqkv = (const float*)d_in[2];   // [3072,1024]
    const float* wo   = (const float*)d_in[3];   // [1024,1024]
    const float* bias = (const float*)d_in[4];   // [H,S,S]
    float* out = (float*)d_out;

    char* w = (char*)d_ws;
    __bf16* x_bf   = (__bf16*)(w);                       // 4M  elems ->  8 MB
    __bf16* wq_bf  = (__bf16*)(w + (8u  << 20));         // 3M  elems ->  6 MB
    __bf16* wo_bf  = (__bf16*)(w + (14u << 20));         // 1M  elems ->  2 MB
    __bf16* q_bf   = (__bf16*)(w + (16u << 20));         // 4M  elems ->  8 MB
    __bf16* k_bf   = (__bf16*)(w + (24u << 20));         // 4M  elems ->  8 MB
    __bf16* vt_bf  = (__bf16*)(w + (32u << 20));         // 4M  elems ->  8 MB
    __bf16* a_bf   = (__bf16*)(w + (40u << 20));         // 4M  elems ->  8 MB

    const int nX = B_ * S_ * D_;          // 4194304
    const int nQ = O3_ * D_;              // 3145728
    const int nO = D_ * (H_ * HD_);       // 1048576

    cvt_bf16<<<(nX + 255) / 256, 256, 0, stream>>>(x,    x_bf,  nX);
    cvt_bf16<<<(nQ + 255) / 256, 256, 0, stream>>>(wqkv, wq_bf, nQ);
    cvt_bf16<<<(nO + 255) / 256, 256, 0, stream>>>(wo,   wo_bf, nO);

    // 128 M-tiles * 48 N4-tiles = 6144 wave tasks / 4 waves per block
    qkv_gemm<<<1536, 128, 0, stream>>>(x_bf, wq_bf, q_bf, k_bf, vt_bf);

    // B*H*(S/16) = 4096 wave tasks / 4 waves per block
    attn<<<1024, 128, 0, stream>>>(q_bf, k_bf, vt_bf, mask, bias, a_bf);

    // 128 M-tiles * 16 N4-tiles = 2048 wave tasks / 4 waves per block
    oproj_gemm<<<512, 128, 0, stream>>>(a_bf, wo_bf, out);
}
